// TSMambaBlock_69140383531163
// MI455X (gfx1250) — compile-verified, matrix operand order checked by reading, not run on hardware
//
#include <hip/hip_runtime.h>
#include <hip/hip_bf16.h>

// ---------------------------------------------------------------------------
// TSMambaBlock for MI455X (gfx1250): WMMA-bf16 GEMMs (pre-packed B fragments)
// + register-state selective scan.
//
// Shapes: D_MODEL=64, D_INNER=256, D_STATE=16, D_CONV=4, DT_RANK=4, N_LAYER=2
// time:  Bsz=256, L=256;  freq: Bsz=512, L=128;  M = Bsz*L = 65536 everywhere.
// ---------------------------------------------------------------------------

#define MTOK   65536         // tokens per phase
#define DMODEL 64
#define DINNER 256
#define DSTATE 16
#define XDB_LD 64            // 36 useful cols, zero-padded to 64

typedef __attribute__((ext_vector_type(16))) __bf16 v16bf;
typedef __attribute__((ext_vector_type(8)))  float  v8f;

__device__ __forceinline__ float siluf(float v)     { return v / (1.f + __expf(-v)); }
__device__ __forceinline__ float softplusf(float v) { return (v > 20.f) ? v : __logf(1.f + __expf(v)); }

// ---------------------------------------------------------------------------
// Weight pack: fp32 weights -> bf16 fragments in WMMA B-operand lane order.
// Packed layout: Wp[(((kt*numNT + nt)*32 + lane)*16) + e]
//   lane = 16*khalf + n%16 ; e<8 -> K = kt*32 + khalf*8 + e
//                            e>=8 -> K = kt*32 + 16 + khalf*8 + (e-8)
//   wKM=0: source W is (N,K) row-major (einsum 'ed' weights)
//   wKM=1: source W is (K,N) row-major (tlin/flin 'eo' weights)
// Rows n >= Nreal are zero-filled (x_proj N=36 -> 64 padding).
// ---------------------------------------------------------------------------
__global__ void k_packW(const float* __restrict__ W, __bf16* __restrict__ Wp,
                        int ldw, int wKM, int Nreal, int numNT, int total)
{
    int i = blockIdx.x * blockDim.x + threadIdx.x;
    if (i >= total) return;
    int e    = i & 15;
    int lane = (i >> 4) & 31;
    int nt   = (i >> 9) % numNT;
    int kt   = (i >> 9) / numNT;
    int n    = nt * 16 + (lane & 15);
    int kb   = (lane >> 4) * 8;
    int ke   = (e < 8) ? (kb + e) : (8 + kb + e);   // e>=8: 16 + kb + (e-8)
    int k    = kt * 32 + ke;
    float v  = 0.f;
    if (n < Nreal) v = wKM ? W[(size_t)k * ldw + n] : W[(size_t)n * ldw + k];
    Wp[i] = (__bf16)v;
}

// ---------------------------------------------------------------------------
// GEMM: C[M,N] = A[M,K] * B (+bias)(+addsrc); A fp32 row-major (cvt in-reg),
// B pre-packed bf16 fragments, f32 WMMA accumulate.
// One wave computes a 16(M) x 64(N) strip: 1 A fragment feeds 4 WMMA tiles.
// All 4 B fragments are issued before the first WMMA so the backend can use
// partial s_wait_loadcnt and overlap matrix ops with in-flight loads.
// Block = 4 waves = 64 rows.  Grid: (M/64, N/64).  K%32==0, N%64==0.
// ---------------------------------------------------------------------------
__global__ void __launch_bounds__(128)
k_gemm(const float* __restrict__ A, int lda,
       const __bf16* __restrict__ Wp, int numNT,
       float* __restrict__ C, int ldc,
       const float* __restrict__ bias,
       const float* __restrict__ addsrc, int ldadd,
       int K)
{
    union AB { v16bf v; __bf16 e[16]; };
    union CF { v8f  v; float  e[8]; };

    const int lane   = threadIdx.x & 31;
    const int wave   = threadIdx.x >> 5;
    const int row0   = (blockIdx.x * 4 + wave) * 16;
    const int ntBase = blockIdx.y * 4;
    const int half   = lane >> 4;        // 0: K 0-7/16-23, 1: K 8-15/24-31
    const int kb     = half * 8;
    const int nl     = lane & 15;
    const int mr     = row0 + nl;

    CF acc[4];
#pragma unroll
    for (int t = 0; t < 4; ++t)
#pragma unroll
        for (int j = 0; j < 8; ++j) acc[t].e[j] = 0.f;

    const int KT = K >> 5;
#pragma unroll 2
    for (int kt = 0; kt < KT; ++kt) {
        const float* Ar = A + (size_t)mr * lda + kt * 32;
        if (kt + 1 < KT) __builtin_prefetch(Ar + 32, 0, 1);   // global_prefetch_b8

        AB a;
#pragma unroll
        for (int e = 0; e < 8; ++e) {
            a.e[e]     = (__bf16)Ar[kb + e];
            a.e[8 + e] = (__bf16)Ar[16 + kb + e];
        }

        // Issue all four B-fragment loads (8x b128) before any WMMA.
        const __bf16* Wk = Wp + (((size_t)kt * numNT + ntBase) * 32 + lane) * 16;
        AB b[4];
#pragma unroll
        for (int t = 0; t < 4; ++t)
            b[t].v = *(const v16bf*)(Wk + (size_t)t * 512);   // 32B vector load

#pragma unroll
        for (int t = 0; t < 4; ++t)
            acc[t].v = __builtin_amdgcn_wmma_f32_16x16x32_bf16(
                false, a.v, false, b[t].v, (short)0, acc[t].v, false, false);
    }

#pragma unroll
    for (int t = 0; t < 4; ++t) {
        int nc = (ntBase + t) * 16 + nl;
        float bv = bias ? bias[nc] : 0.f;
#pragma unroll
        for (int j = 0; j < 8; ++j) {
            int mrow = row0 + j + half * 8;   // C layout: VGPR j -> M=j (+8 hi half)
            float v = acc[t].e[j] + bv;
            if (addsrc) v += addsrc[(size_t)mrow * ldadd + nc];
            C[(size_t)mrow * ldc + nc] = v;
        }
    }
}

// ---------------------------------------------------------------------------
// Elementwise / reshape kernels
// ---------------------------------------------------------------------------
__global__ void k_zero(float* __restrict__ p, int n) {
    int i = blockIdx.x * blockDim.x + threadIdx.x;
    if (i < n) p[i] = 0.f;
}

__global__ void k_axpy(float* __restrict__ resi, const float* __restrict__ src, int n) {
    int i = blockIdx.x * blockDim.x + threadIdx.x;
    if (i < n) resi[i] = src[i] + resi[i];
}

__global__ void k_rmsnorm(const float* __restrict__ resi, const float* __restrict__ w,
                          float* __restrict__ u) {
    int m = blockIdx.x * blockDim.x + threadIdx.x;   // one token per thread
    const float* r = resi + (size_t)m * DMODEL;
    float ss = 0.f;
#pragma unroll 8
    for (int c = 0; c < DMODEL; ++c) { float v = r[c]; ss += v * v; }
    float sc = rsqrtf(ss * (1.f / DMODEL) + 1e-5f);
    float* o = u + (size_t)m * DMODEL;
#pragma unroll 8
    for (int c = 0; c < DMODEL; ++c) o[c] = r[c] * sc * w[c];
}

__global__ void k_conv_silu(const float* __restrict__ XZ, const float* __restrict__ cw,
                            const float* __restrict__ cb, float* __restrict__ XSILU, int Lseq) {
    int i = blockIdx.x * blockDim.x + threadIdx.x;   // (m, d)
    int d = i & (DINNER - 1);
    int m = i >> 8;
    int t = m % Lseq;
    float acc = cb[d];
#pragma unroll
    for (int k = 0; k < 4; ++k) {
        int tk = t + k - 3;
        if (tk >= 0) acc += XZ[(size_t)(m + k - 3) * 512 + d] * cw[d * 4 + k];
    }
    XSILU[(size_t)m * DINNER + d] = siluf(acc);
}

// Selective scan: one lane per (sequence, channel); 16-state in VGPRs;
// shared per-token (dt_r, B, C) staged through LDS in 16-step chunks.
__global__ void __launch_bounds__(256)
k_scan(const float* __restrict__ XDB, const float* __restrict__ XSILU,
       const float* __restrict__ A_log, const float* __restrict__ dtw,
       const float* __restrict__ dtb, float* __restrict__ YS, int Lseq)
{
    __shared__ float sm[16 * 36];
    const int seq = blockIdx.x;
    const int d   = threadIdx.x;

    float A[DSTATE];
#pragma unroll
    for (int s = 0; s < DSTATE; ++s) A[s] = -__expf(A_log[d * DSTATE + s]);
    const float w0 = dtw[d * 4 + 0], w1 = dtw[d * 4 + 1];
    const float w2 = dtw[d * 4 + 2], w3 = dtw[d * 4 + 3];
    const float bd = dtb[d];

    float h[DSTATE];
#pragma unroll
    for (int s = 0; s < DSTATE; ++s) h[s] = 0.f;

    for (int t0 = 0; t0 < Lseq; t0 += 16) {
        __syncthreads();
        for (int i = threadIdx.x; i < 16 * 36; i += 256) {
            int row = i / 36, col = i - row * 36;
            sm[i] = XDB[(size_t)(seq * Lseq + t0 + row) * XDB_LD + col];
        }
        __syncthreads();
#pragma unroll 4
        for (int tt = 0; tt < 16; ++tt) {
            const float* sp = &sm[tt * 36];
            size_t m = (size_t)seq * Lseq + t0 + tt;
            float dt  = softplusf(sp[0] * w0 + sp[1] * w1 + sp[2] * w2 + sp[3] * w3 + bd);
            float dtx = dt * XSILU[m * DINNER + d];
            float y = 0.f;
#pragma unroll
            for (int s = 0; s < DSTATE; ++s) {
                h[s] = __expf(dt * A[s]) * h[s] + dtx * sp[4 + s];
                y += h[s] * sp[20 + s];
            }
            YS[m * DINNER + d] = y;
        }
    }
}

__global__ void k_gate(float* __restrict__ YS, const float* __restrict__ XSILU,
                       const float* __restrict__ XZ, const float* __restrict__ Dp) {
    int i = blockIdx.x * blockDim.x + threadIdx.x;
    int d = i & (DINNER - 1);
    size_t m = (size_t)(i >> 8);
    float z = XZ[m * 512 + 256 + d];
    YS[m * DINNER + d] = (YS[m * DINNER + d] + XSILU[m * DINNER + d] * Dp[d]) * siluf(z);
}

__global__ void k_flip(float* __restrict__ dst, const float* __restrict__ src, int Lseq) {
    int i = blockIdx.x * blockDim.x + threadIdx.x;    // (m, c)
    int c = i & (DMODEL - 1);
    int m = i >> 6;
    int seq = m / Lseq, t = m - seq * Lseq;
    dst[(size_t)(seq * Lseq + (Lseq - 1 - t)) * DMODEL + c] = src[(size_t)m * DMODEL + c];
}

// Y2[:, off:off+64] = X + RESI  (optionally flipped back along t)
__global__ void k_stack_out(float* __restrict__ Y2, const float* __restrict__ X,
                            const float* __restrict__ RESI, int off, int flip, int Lseq) {
    int i = blockIdx.x * blockDim.x + threadIdx.x;
    int c = i & (DMODEL - 1);
    int m = i >> 6;
    int seq = m / Lseq, t = m - seq * Lseq;
    int to = flip ? (Lseq - 1 - t) : t;
    Y2[(size_t)(seq * Lseq + to) * 128 + off + c] =
        X[(size_t)m * DMODEL + c] + RESI[(size_t)m * DMODEL + c];
}

// x(b,c,t,f) -> H(b*f, t, c)
__global__ void k_reshape_in(float* __restrict__ H, const float* __restrict__ x) {
    int i = blockIdx.x * blockDim.x + threadIdx.x;    // H linear index
    int c = i & 63;
    int r = i >> 6;
    int t = r & 255;  r >>= 8;
    int f = r & 127;
    int b = r >> 7;
    H[i] = x[(((size_t)(b * 64 + c) * 256 + t) * 128) + f];
}

// H2(b*f, t, c) -> Hf(b*t, f, c)
__global__ void k_reshape_t2f(float* __restrict__ Hf, const float* __restrict__ H2) {
    int i = blockIdx.x * blockDim.x + threadIdx.x;    // Hf linear index
    int c = i & 63;
    int r = i >> 6;
    int f = r & 127; r >>= 7;
    int t = r & 255;
    int b = r >> 8;
    Hf[i] = H2[(((size_t)(b * 128 + f) * 256 + t) * 64) + c];
}

// Hf2(b*t, f, c) -> out(b,c,t,f)
__global__ void k_reshape_out(float* __restrict__ out, const float* __restrict__ Hf2) {
    int i = blockIdx.x * blockDim.x + threadIdx.x;    // out linear index
    int f = i & 127;
    int r = i >> 7;
    int t = r & 255; r >>= 8;
    int c = r & 63;
    int b = r >> 6;
    out[i] = Hf2[(((size_t)(b * 256 + t) * 128 + f) * 64) + c];
}

// ---------------------------------------------------------------------------
// Host orchestration
// ---------------------------------------------------------------------------
struct LayerP {
    const float *A_log, *D, *conv_b, *conv_w, *dt_b, *dt_w, *in_w, *norm_w, *out_w, *x_w;
};

static LayerP layer_at(void* const* d_in, int base) {
    LayerP p;
    p.A_log  = (const float*)d_in[base + 0];
    p.D      = (const float*)d_in[base + 1];
    p.conv_b = (const float*)d_in[base + 2];
    p.conv_w = (const float*)d_in[base + 3];
    p.dt_b   = (const float*)d_in[base + 4];
    p.dt_w   = (const float*)d_in[base + 5];
    p.in_w   = (const float*)d_in[base + 6];
    p.norm_w = (const float*)d_in[base + 7];
    p.out_w  = (const float*)d_in[base + 8];
    p.x_w    = (const float*)d_in[base + 9];
    return p;
}

struct Bufs {
    float *H, *XF, *RESI, *U, *X, *Y2, *H2, *XZ, *XSILU, *XDB, *YS;
    __bf16 *Wp;
};

// pack + gemm helper: N must be padded multiple of 64 (Npad), source N = Nreal
static void pack_gemm(const float* A, int lda, const float* W, int ldw, int wKM,
                      int Nreal, int Npad, int K,
                      float* C, int ldc, const float* bias,
                      const float* addsrc, int ldadd, Bufs& B, hipStream_t s) {
    const int numNT = Npad / 16;
    const int total = (K / 32) * numNT * 512;   // == K * Npad
    k_packW<<<(total + 255) / 256, 256, 0, s>>>(W, B.Wp, ldw, wKM, Nreal, numNT, total);
    k_gemm<<<dim3(MTOK / 64, Npad / 64), dim3(128), 0, s>>>(
        A, lda, B.Wp, numNT, C, ldc, bias, addsrc, ldadd, K);
}

static void run_mamba(const LayerP& p, Bufs& B, float* Xout, int Bsz, int Lseq, hipStream_t s) {
    const int M = MTOK;
    // in_proj: XZ = U @ in_w^T   (M x 512, K=64)
    pack_gemm(B.U, DMODEL, p.in_w, DMODEL, 0, 512, 512, DMODEL,
              B.XZ, 512, nullptr, nullptr, 0, B, s);
    k_conv_silu<<<(M * DINNER) / 256, 256, 0, s>>>(B.XZ, p.conv_w, p.conv_b, B.XSILU, Lseq);
    // x_proj: XDB = XSILU @ x_w^T   (M x 36 zero-padded to 64, K=256)
    pack_gemm(B.XSILU, DINNER, p.x_w, DINNER, 0, 36, 64, DINNER,
              B.XDB, XDB_LD, nullptr, nullptr, 0, B, s);
    k_scan<<<Bsz, 256, 0, s>>>(B.XDB, B.XSILU, p.A_log, p.dt_w, p.dt_b, B.YS, Lseq);
    k_gate<<<(M * DINNER) / 256, 256, 0, s>>>(B.YS, B.XSILU, B.XZ, p.D);
    // out_proj: X = YS @ out_w^T   (M x 64, K=256)
    pack_gemm(B.YS, DINNER, p.out_w, DINNER, 0, 64, 64, DINNER,
              Xout, DMODEL, nullptr, nullptr, 0, B, s);
}

static void run_stack(const LayerP* lyr, const float* Xin, float* Y2, int coloff, int flip,
                      Bufs& B, int Bsz, int Lseq, hipStream_t s) {
    const int n64 = MTOK * DMODEL;
    k_zero<<<n64 / 256, 256, 0, s>>>(B.RESI, n64);
    for (int i = 0; i < 2; ++i) {
        const float* src = (i == 0) ? Xin : B.X;
        k_axpy<<<n64 / 256, 256, 0, s>>>(B.RESI, src, n64);
        k_rmsnorm<<<MTOK / 256, 256, 0, s>>>(B.RESI, lyr[i].norm_w, B.U);
        run_mamba(lyr[i], B, B.X, Bsz, Lseq, s);
    }
    k_stack_out<<<n64 / 256, 256, 0, s>>>(Y2, B.X, B.RESI, coloff, flip, Lseq);
}

static void run_bimamba(const LayerP* fwd, const LayerP* bwd, float* H, Bufs& B,
                        int Bsz, int Lseq, hipStream_t s) {
    const int n64 = MTOK * DMODEL;
    run_stack(fwd, H, B.Y2, 0, 0, B, Bsz, Lseq, s);
    k_flip<<<n64 / 256, 256, 0, s>>>(B.XF, H, Lseq);
    run_stack(bwd, B.XF, B.Y2, 64, 1, B, Bsz, Lseq, s);
}

extern "C" void kernel_launch(void* const* d_in, const int* in_sizes, int n_in,
                              void* d_out, int out_size, void* d_ws, size_t ws_size,
                              hipStream_t stream) {
    (void)in_sizes; (void)out_size; (void)ws_size;
    if (n_in < 85) return;

    // JAX pytree (sorted-key) flattening after 'x':
    // 0:x  1:flin_b  2:flin_w  3..:freq{bwd0,bwd1,fwd0,fwd1} 43..:time{bwd0,bwd1,fwd0,fwd1}
    // 83:tlin_b 84:tlin_w ; each layer: A_log,D,conv_b,conv_w,dt_proj_b,dt_proj_w,
    //                                   in_proj_w,norm_w,out_proj_w,x_proj_w
    const float* x      = (const float*)d_in[0];
    const float* flin_b = (const float*)d_in[1];
    const float* flin_w = (const float*)d_in[2];
    const float* tlin_b = (const float*)d_in[83];
    const float* tlin_w = (const float*)d_in[84];
    LayerP freq_bwd[2] = { layer_at(d_in, 3),  layer_at(d_in, 13) };
    LayerP freq_fwd[2] = { layer_at(d_in, 23), layer_at(d_in, 33) };
    LayerP time_bwd[2] = { layer_at(d_in, 43), layer_at(d_in, 53) };
    LayerP time_fwd[2] = { layer_at(d_in, 63), layer_at(d_in, 73) };

    // Workspace carve-out
    char* w = (char*)d_ws;
    size_t off = 0;
    auto carve = [&](size_t bytes) -> char* {
        char* p = w + off;
        off += (bytes + 255) & ~(size_t)255;
        return p;
    };
    Bufs B;
    const size_t M = MTOK;
    B.H     = (float*)carve(M * 64  * 4);
    B.XF    = (float*)carve(M * 64  * 4);
    B.RESI  = (float*)carve(M * 64  * 4);
    B.U     = (float*)carve(M * 64  * 4);
    B.X     = (float*)carve(M * 64  * 4);
    B.H2    = (float*)carve(M * 64  * 4);
    B.Y2    = (float*)carve(M * 128 * 4);
    B.XZ    = (float*)carve(M * 512 * 4);
    B.XSILU = (float*)carve(M * 256 * 4);
    B.XDB   = (float*)carve(M * XDB_LD * 4);
    B.YS    = (float*)carve(M * 256 * 4);
    B.Wp    = (__bf16*)carve(64 * 512 * 2);   // largest packed weight (in_proj)

    const int nElem = MTOK * DMODEL;     // 4,194,304

    // ---- time phase: (b*f, t=256, c) ----
    k_reshape_in<<<nElem / 256, 256, 0, stream>>>(B.H, x);
    run_bimamba(time_fwd, time_bwd, B.H, B, 256, 256, stream);
    // H2 = Y2 @ tlin_w (K-major 128x64) + tlin_b + H
    pack_gemm(B.Y2, 128, tlin_w, 64, 1, 64, 64, 128,
              B.H2, 64, tlin_b, B.H, 64, B, stream);

    // ---- freq phase: (b*t, f=128, c) ----
    k_reshape_t2f<<<nElem / 256, 256, 0, stream>>>(B.H, B.H2);
    run_bimamba(freq_fwd, freq_bwd, B.H, B, 512, 128, stream);
    pack_gemm(B.Y2, 128, flin_w, 64, 1, 64, 64, 128,
              B.H2, 64, flin_b, B.H, 64, B, stream);

    // ---- final layout: (b, c, t, f) ----
    k_reshape_out<<<nElem / 256, 256, 0, stream>>>((float*)d_out, B.H2);
}
